// TVNet_ctrl_62672162783478
// MI455X (gfx1250) — matrix operand order repaired and checked by reading, not compiled, not run on hardware
//
// TVNet ADMM solver for MI455X (gfx1250, CDNA5).
// FFT solve implemented as bf16 DFT-GEMMs on v_wmma_f32_16x16x32_bf16:
// the 512x512 DFT cos/-sin tables (0.5MB bf16 each, symmetric) stay hot in
// the 192MB L2, so the 4 transform passes/iteration become L2-resident GEMMs
// (~1.6 TFLOP bf16 total) instead of a scalar-VALU FFT, matching the
// ~0.5ms HBM roofline (10 x ~1GB / 23.3 TB/s) while using the matrix cores.
// All GEMM operands stream as contiguous b128 loads: the column-pass inputs
// (Y, Zh) are written TRANSPOSED by their producers (1 scattered 4B store per
// element) instead of being gather-loaded 32x by the consumers.

#include <hip/hip_runtime.h>
#include <math.h>
#include <cstddef>

#define BATCH 16
#define CCH   3
#define GCH   6
#define HW    512
#define NN    (HW * HW)          // 262144 = 2^18
#define NIMG  (BATCH * CCH)      // 48
#define NIMG6 (BATCH * GCH)      // 96
#define TVNITER 10

typedef __attribute__((ext_vector_type(16))) __bf16        v16bf;
typedef __attribute__((ext_vector_type(8)))  float         v8f;
typedef __attribute__((ext_vector_type(4)))  unsigned int  u32x4;

union BFrag {
    v16bf          v;
    unsigned short u[16];
    u32x4          q[2];
};

static __device__ inline unsigned short f2bf(float f) {
    unsigned int x = __float_as_uint(f);
    unsigned int r = ((x >> 16) & 1u) + 0x7FFFu;   // round to nearest even
    return (unsigned short)((x + r) >> 16);
}

static __device__ inline v8f zero8() {
    v8f z = {0.f, 0.f, 0.f, 0.f, 0.f, 0.f, 0.f, 0.f};
    return z;
}

static __device__ inline v8f wmma_bf16(v16bf a, v16bf b, v8f c) {
    // D = A(16x32 bf16) x B(32x16 bf16) + C(16x16 f32)
    return __builtin_amdgcn_wmma_f32_16x16x32_bf16(
        /*neg_a=*/false, a, /*neg_b=*/false, b,
        /*c_mod=*/(short)0, c, /*reuse_a=*/false, /*reuse_b=*/false);
}

// ---- fragment loaders (CDNA5 ISA 7.12.2 layouts) --------------------------
// A (16x32, 16-bit): lane L holds row M = L&15; elems 0..7 -> K = kb+half*8+j,
// elems 8..15 -> K = kb+16+half*8+j, half = L>>4.  Two contiguous 16B loads.
static __device__ inline v16bf loadA_bf(const unsigned short* __restrict__ M,
                                        int mBase, int kBase, int lane) {
    int half = lane >> 4;
    const unsigned short* row = M + (size_t)(mBase + (lane & 15)) * HW;
    BFrag f;
    f.q[0] = *(const u32x4*)(row + kBase + half * 8);
    f.q[1] = *(const u32x4*)(row + kBase + 16 + half * 8);
    return f.v;
}

// B (32x16, 16-bit): lane L holds column N = nBase + (L&15); elems 0..7 -> K =
// kBase+half*8+j, elems 8..15 -> K = kBase+16+half*8+j.  The DFT tables are SYMMETRIC, so column n of T is
// row n of T -> two contiguous 16B loads per fragment.
static __device__ inline v16bf loadB_sym(const unsigned short* __restrict__ T,
                                         int kBase, int nBase, int lane) {
    int half = lane >> 4;
    const unsigned short* row =
        T + (size_t)(nBase + (lane & 15)) * HW + kBase + half * 8;
    BFrag f;
    f.q[0] = *(const u32x4*)(row);
    f.q[1] = *(const u32x4*)(row + 16);
    return f.v;
}

// B from a TRANSPOSED interleaved bf16 complex buffer Mt[n][k] (re lo16,
// im hi16): the 16 K-elements a lane needs are two runs of 8 consecutive u32 (K and K+16) -> 4xb128.
static __device__ inline void loadB_cplxT(const unsigned int* __restrict__ Mt,
                                          int kBase, int nBase, int lane,
                                          v16bf* bre, v16bf* bim) {
    int half = lane >> 4;
    const unsigned int* p =
        Mt + (size_t)(nBase + (lane & 15)) * HW + kBase + half * 8;
    __builtin_prefetch(p + 32, 0, 1);     // next k-block of this row
    union { u32x4 q[4]; unsigned int w[16]; } d;
    d.q[0] = *(const u32x4*)(p);
    d.q[1] = *(const u32x4*)(p + 4);
    d.q[2] = *(const u32x4*)(p + 16);
    d.q[3] = *(const u32x4*)(p + 20);
    BFrag r, i;
#pragma unroll
    for (int e = 0; e < 16; ++e) {
        r.u[e] = (unsigned short)(d.w[e] & 0xFFFFu);
        i.u[e] = (unsigned short)(d.w[e] >> 16);
    }
    *bre = r.v;
    *bim = i.v;
}

// A from interleaved bf16 complex buffer, row-major: 4xb128 per fragment pair.
static __device__ inline void loadA_cplx(const unsigned int* __restrict__ Mc,
                                         int mBase, int kBase, int lane,
                                         v16bf* are, v16bf* aim) {
    int half = lane >> 4;
    const unsigned int* row = Mc + (size_t)(mBase + (lane & 15)) * HW;
    union { u32x4 q[2]; unsigned int w[8]; } d0, d1;
    d0.q[0] = *(const u32x4*)(row + kBase + half * 8);
    d0.q[1] = *(const u32x4*)(row + kBase + half * 8 + 4);
    d1.q[0] = *(const u32x4*)(row + kBase + 16 + half * 8);
    d1.q[1] = *(const u32x4*)(row + kBase + 16 + half * 8 + 4);
    BFrag r, i;
#pragma unroll
    for (int j = 0; j < 8; ++j) {
        r.u[j]     = (unsigned short)(d0.w[j] & 0xFFFFu);
        i.u[j]     = (unsigned short)(d0.w[j] >> 16);
        r.u[j + 8] = (unsigned short)(d1.w[j] & 0xFFFFu);
        i.u[j + 8] = (unsigned short)(d1.w[j] >> 16);
    }
    *are = r.v;
    *aim = i.v;
}

// ---- one-time setup kernels ----------------------------------------------

// bf16 DFT tables: Fre[k][n] = cos(2*pi*k*n/512), Fim[k][n] = -sin(...)
__global__ void tvnet_fill_tables(unsigned short* __restrict__ Fre,
                                  unsigned short* __restrict__ Fim) {
    int idx = blockIdx.x * blockDim.x + threadIdx.x;
    if (idx >= NN) return;
    int k = idx >> 9, n = idx & 511;
    int prod = (k * n) & 511;
    float ang = (float)prod * (6.283185307179586f / 512.0f);
    float s, c;
    sincosf(ang, &s, &c);
    Fre[idx] = f2bf(c);
    Fim[idx] = f2bf(-s);
}

// kerW = rho * flip(kerK); kerH = I + rho * sum_kO autocorr(kerK)  (tiny)
__global__ void tvnet_prep(const float* __restrict__ kerK,
                           const float* __restrict__ rho,
                           float* __restrict__ kerW,
                           float* __restrict__ kerH) {
    if (threadIdx.x != 0 || blockIdx.x != 0) return;
    for (int g = 0; g < GCH; ++g)
        for (int dy = 0; dy < 3; ++dy)
            for (int dx = 0; dx < 3; ++dx)
                kerW[g * 9 + dy * 3 + dx] =
                    rho[g] * kerK[g * 9 + (2 - dy) * 3 + (2 - dx)];
    for (int ch = 0; ch < CCH; ++ch)
        for (int a = 0; a < 5; ++a)
            for (int b = 0; b < 5; ++b) {
                float s = 0.f;
                for (int j = 0; j < 2; ++j) {
                    int g = ch * 2 + j;
                    float ac = 0.f;
                    for (int dy = 0; dy < 3; ++dy)
                        for (int dx = 0; dx < 3; ++dx) {
                            int ya = dy + a - 2, xb = dx + b - 2;
                            if (ya >= 0 && ya < 3 && xb >= 0 && xb < 3)
                                ac += kerK[g * 9 + ya * 3 + xb] *
                                      kerK[g * 9 + dy * 3 + dx];
                        }
                    s += rho[g] * ac;
                }
                kerH[ch * 25 + a * 5 + b] =
                    s + ((a == 2 && b == 2) ? 1.0f : 0.0f);
            }
}

// invOTF = conj(otf)/(|otf|^2) / 512^2   (both ifft 1/N scalings folded in)
__global__ void tvnet_invotf(const float* __restrict__ kerH,
                             float* __restrict__ inv) {
    int idx = blockIdx.x * blockDim.x + threadIdx.x;
    if (idx >= CCH * NN) return;
    int ch  = idx >> 18;
    int pix = idx & (NN - 1);
    int u = pix >> 9, v = pix & 511;
    float orr = 0.f, oii = 0.f;
    for (int a = 0; a < 5; ++a)
        for (int b = 0; b < 5; ++b) {
            float w   = kerH[ch * 25 + a * 5 + b];
            float ang = -(6.283185307179586f / 512.0f) *
                        (float)(u * (a - 2) + v * (b - 2));
            float s, c;
            sincosf(ang, &s, &c);
            orr += w * c;
            oii += w * s;
        }
    float d = (orr * orr + oii * oii) * 262144.0f;
    float r = 1.0f / d;
    inv[2 * idx]     = orr * r;
    inv[2 * idx + 1] = -oii * r;
}

// ---- per-iteration kernels ------------------------------------------------

// w1 = f + sum_kO conv3x3_circ(p - mu, kerW)   -> bf16 (GEMM A operand)
__global__ void tvnet_w1(const float* __restrict__ f,
                         const float* __restrict__ p,
                         const float* __restrict__ mu,
                         const float* __restrict__ kerW,
                         unsigned short* __restrict__ w1) {
    int idx = blockIdx.x * blockDim.x + threadIdx.x;
    if (idx >= NIMG * NN) return;
    int img = idx >> 18;
    int pix = idx & (NN - 1);
    int y = pix >> 9, x = pix & 511;
    int b = img / CCH, ch = img - b * CCH;
    float acc = f[idx];
#pragma unroll
    for (int j = 0; j < 2; ++j) {
        int g = ch * 2 + j;
        const float* pp = p  + ((size_t)(b * GCH + g) << 18);
        const float* mm = mu + ((size_t)(b * GCH + g) << 18);
        const float* kw = kerW + g * 9;
#pragma unroll
        for (int dy = 0; dy < 3; ++dy) {
            int ro = ((y + dy + 511) & 511) << 9;
#pragma unroll
            for (int dx = 0; dx < 3; ++dx) {
                int xx = (x + dx + 511) & 511;
                acc += kw[dy * 3 + dx] * (pp[ro + xx] - mm[ro + xx]);
            }
        }
    }
    w1[idx] = f2bf(acc);
}

// Yt[kx][r] = (w1 * F)[r][kx]  (forward DFT along rows; TRANSPOSED output so
// the column pass streams contiguous b128 B-fragments)
__global__ void __launch_bounds__(256)
tvnet_rowfft(const unsigned short* __restrict__ w1,
             const unsigned short* __restrict__ Fre,
             const unsigned short* __restrict__ Fim,
             unsigned int* __restrict__ Yt) {
    int wave = threadIdx.x >> 5, lane = threadIdx.x & 31;
    int flat = blockIdx.x * 8 + wave;
    int img  = flat >> 10;
    int t    = flat & 1023;
    int tM = (t >> 5) << 4, tN = (t & 31) << 4;
    const unsigned short* X = w1 + (size_t)img * NN;
    unsigned int* Yp = Yt + (size_t)img * NN;
    v8f aR = zero8(), aI = zero8();
    for (int kb = 0; kb < HW; kb += 32) {
        v16bf a  = loadA_bf(X, tM, kb, lane);
        v16bf br = loadB_sym(Fre, kb, tN, lane);
        v16bf bi = loadB_sym(Fim, kb, tN, lane);
        aR = wmma_bf16(a, br, aR);
        aI = wmma_bf16(a, bi, aI);
    }
    int col = tN + (lane & 15);                 // kx
    int r0  = tM + (lane >> 4) * 8;             // r
#pragma unroll
    for (int v = 0; v < 8; ++v) {
        unsigned int pk = (unsigned int)f2bf(aR[v]) |
                          ((unsigned int)f2bf(aI[v]) << 16);
        Yp[(size_t)col * HW + (r0 + v)] = pk;   // transposed store
    }
}

// Z = F * Y (forward DFT along columns), Zh = Z .* invOTF[channel];
// reads Yt[kx][r] contiguously, writes Zht[kx][ky] transposed.
__global__ void __launch_bounds__(256)
tvnet_colfft_otf(const unsigned int* __restrict__ Yt,
                 const unsigned short* __restrict__ Fre,
                 const unsigned short* __restrict__ Fim,
                 const float* __restrict__ inv,
                 unsigned int* __restrict__ Zht) {
    int wave = threadIdx.x >> 5, lane = threadIdx.x & 31;
    int flat = blockIdx.x * 8 + wave;
    int img  = flat >> 10;
    int t    = flat & 1023;
    int tM = (t >> 5) << 4, tN = (t & 31) << 4;  // tM: ky, tN: kx
    int ch = img - (img / CCH) * CCH;
    const unsigned int* Yp = Yt + (size_t)img * NN;
    unsigned int* Zp = Zht + (size_t)img * NN;
    v8f p1 = zero8(), p2 = zero8(), p3 = zero8(), p4 = zero8();
    for (int kb = 0; kb < HW; kb += 32) {
        v16bf are = loadA_bf(Fre, tM, kb, lane);
        v16bf aim = loadA_bf(Fim, tM, kb, lane);
        v16bf br, bi;
        loadB_cplxT(Yp, kb, tN, lane, &br, &bi);
        p1 = wmma_bf16(are, br, p1);   // Fre*Yr
        p2 = wmma_bf16(aim, bi, p2);   // Fim*Yi
        p3 = wmma_bf16(are, bi, p3);   // Fre*Yi
        p4 = wmma_bf16(aim, br, p4);   // Fim*Yr
    }
    int col = tN + (lane & 15);                  // kx
    int r0  = tM + (lane >> 4) * 8;              // ky
#pragma unroll
    for (int v = 0; v < 8; ++v) {
        float zr = p1[v] - p2[v];
        float zi = p3[v] + p4[v];
        size_t o = ((size_t)ch << 18) + (size_t)(r0 + v) * HW + col;
        float ir = inv[2 * o], ii = inv[2 * o + 1];
        float hr = zr * ir - zi * ii;
        float hi = zr * ii + zi * ir;
        Zp[(size_t)col * HW + (r0 + v)] =        // transposed store
            (unsigned int)f2bf(hr) | ((unsigned int)f2bf(hi) << 16);
    }
}

// V = conj(F) * Zh (inverse DFT along columns; 1/N folded into invOTF);
// reads Zht[kx][ky] contiguously, writes V[r][kx] row-major (coalesced).
__global__ void __launch_bounds__(256)
tvnet_colifft(const unsigned int* __restrict__ Zht,
              const unsigned short* __restrict__ Fre,
              const unsigned short* __restrict__ Fim,
              unsigned int* __restrict__ V) {
    int wave = threadIdx.x >> 5, lane = threadIdx.x & 31;
    int flat = blockIdx.x * 8 + wave;
    int img  = flat >> 10;
    int t    = flat & 1023;
    int tM = (t >> 5) << 4, tN = (t & 31) << 4;  // tM: r, tN: kx
    const unsigned int* Zp = Zht + (size_t)img * NN;
    unsigned int* Vp = V + (size_t)img * NN;
    v8f p1 = zero8(), p2 = zero8(), p3 = zero8(), p4 = zero8();
    for (int kb = 0; kb < HW; kb += 32) {
        v16bf are = loadA_bf(Fre, tM, kb, lane);
        v16bf aim = loadA_bf(Fim, tM, kb, lane);
        v16bf br, bi;
        loadB_cplxT(Zp, kb, tN, lane, &br, &bi);
        p1 = wmma_bf16(are, br, p1);   // Fre*Zr
        p2 = wmma_bf16(aim, bi, p2);   // Fim*Zi
        p3 = wmma_bf16(are, bi, p3);   // Fre*Zi
        p4 = wmma_bf16(aim, br, p4);   // Fim*Zr
    }
    int col = tN + (lane & 15);
    int r0  = tM + (lane >> 4) * 8;
#pragma unroll
    for (int v = 0; v < 8; ++v) {
        float vr = p1[v] + p2[v];      // conj(F): Fre*Zr + Fim*Zi
        float vi = p3[v] - p4[v];      //          Fre*Zi - Fim*Zr
        Vp[(size_t)(r0 + v) * HW + col] =        // row-major store
            (unsigned int)f2bf(vr) | ((unsigned int)f2bf(vi) << 16);
    }
}

// u = Re(V * conj(F)) = Vr*Fre + Vi*Fim  (inverse DFT along rows, real out)
__global__ void __launch_bounds__(256)
tvnet_rowifft(const unsigned int* __restrict__ V,
              const unsigned short* __restrict__ Fre,
              const unsigned short* __restrict__ Fim,
              float* __restrict__ u) {
    int wave = threadIdx.x >> 5, lane = threadIdx.x & 31;
    int flat = blockIdx.x * 8 + wave;
    int img  = flat >> 10;
    int t    = flat & 1023;
    int tM = (t >> 5) << 4, tN = (t & 31) << 4;
    const unsigned int* Vp = V + (size_t)img * NN;
    float* up = u + (size_t)img * NN;
    v8f p1 = zero8(), p2 = zero8();
    for (int kb = 0; kb < HW; kb += 32) {
        v16bf ar, ai;
        loadA_cplx(Vp, tM, kb, lane, &ar, &ai);
        v16bf br = loadB_sym(Fre, kb, tN, lane);
        v16bf bi = loadB_sym(Fim, kb, tN, lane);
        p1 = wmma_bf16(ar, br, p1);
        p2 = wmma_bf16(ai, bi, p2);
    }
    int col = tN + (lane & 15);
    int r0  = tM + (lane >> 4) * 8;
#pragma unroll
    for (int v = 0; v < 8; ++v)
        up[(size_t)(r0 + v) * HW + col] = p1[v] + p2[v];
}

// c = conv3x3_circ(u[ch(g)], kerK[g]);  p = soft(c+mu, gamma);  mu += c - p
__global__ void tvnet_prox(const float* __restrict__ u,
                           const float* __restrict__ kerK,
                           const float* __restrict__ gamma,
                           float* __restrict__ p,
                           float* __restrict__ mu) {
    int idx = blockIdx.x * blockDim.x + threadIdx.x;
    if (idx >= NIMG6 * NN) return;
    int img = idx >> 18;
    int pix = idx & (NN - 1);
    int y = pix >> 9, x = pix & 511;
    int b = img / GCH, g = img - b * GCH;
    int ch = g >> 1;                        // repeat(u, KO=2)
    const float* uc = u + ((size_t)(b * CCH + ch) << 18);
    const float* kk = kerK + g * 9;
    float c = 0.f;
#pragma unroll
    for (int dy = 0; dy < 3; ++dy) {
        int ro = ((y + dy + 511) & 511) << 9;
#pragma unroll
        for (int dx = 0; dx < 3; ++dx)
            c += kk[dy * 3 + dx] * uc[ro + ((x + dx + 511) & 511)];
    }
    float tv = c + mu[idx];
    float a  = fabsf(tv) - gamma[g];
    a = a > 0.f ? a : 0.f;
    float pn = copysignf(a, tv);
    p[idx]  = pn;
    mu[idx] = tv - pn;
}

// ---- driver ---------------------------------------------------------------
extern "C" void kernel_launch(void* const* d_in, const int* in_sizes, int n_in,
                              void* d_out, int out_size, void* d_ws,
                              size_t ws_size, hipStream_t stream) {
    (void)in_sizes; (void)n_in; (void)out_size; (void)ws_size;
    const float* x     = (const float*)d_in[0];   // (16,3,512,512)
    const float* kerK  = (const float*)d_in[1];   // (6,1,3,3)
    const float* gamma = (const float*)d_in[2];   // (1,6,1,1)
    const float* rho   = (const float*)d_in[3];   // (6,1,1,1)
    float* uout = (float*)d_out;

    char* base = (char*)d_ws;
    size_t off = 0;
    auto take = [&](size_t bytes) {
        char* ptr = base + off;
        off = (off + bytes + 255) & ~(size_t)255;
        return ptr;
    };
    float*          p_  = (float*)take((size_t)NIMG6 * NN * 4);         // 100.7MB
    float*          mu_ = (float*)take((size_t)NIMG6 * NN * 4);         // 100.7MB
    unsigned short* w1  = (unsigned short*)take((size_t)NIMG * NN * 2); //  25.2MB
    unsigned int*   Yt  = (unsigned int*)take((size_t)NIMG * NN * 4);   //  50.3MB
    unsigned int*   Zht = (unsigned int*)take((size_t)NIMG * NN * 4);   //  50.3MB
    unsigned int*   V   = (unsigned int*)take((size_t)NIMG * NN * 4);   //  50.3MB
    unsigned short* Fre = (unsigned short*)take((size_t)NN * 2);
    unsigned short* Fim = (unsigned short*)take((size_t)NN * 2);
    float*          inv = (float*)take((size_t)CCH * NN * 8);           //   6.3MB
    float*          kerW = (float*)take(64 * sizeof(float));
    float*          kerH = (float*)take(128 * sizeof(float));

    hipMemsetAsync(p_,  0, (size_t)NIMG6 * NN * 4, stream);
    hipMemsetAsync(mu_, 0, (size_t)NIMG6 * NN * 4, stream);

    tvnet_fill_tables<<<NN / 256, 256, 0, stream>>>(Fre, Fim);
    tvnet_prep<<<1, 32, 0, stream>>>(kerK, rho, kerW, kerH);
    tvnet_invotf<<<(CCH * NN) / 256, 256, 0, stream>>>(kerH, inv);

    const int gemmBlocks = NIMG * 1024 / 8;   // 16x16 tile per wave, 8 waves/blk
    for (int it = 0; it < TVNITER; ++it) {
        tvnet_w1<<<(NIMG * NN) / 256, 256, 0, stream>>>(x, p_, mu_, kerW, w1);
        tvnet_rowfft<<<gemmBlocks, 256, 0, stream>>>(w1, Fre, Fim, Yt);
        tvnet_colfft_otf<<<gemmBlocks, 256, 0, stream>>>(Yt, Fre, Fim, inv, Zht);
        tvnet_colifft<<<gemmBlocks, 256, 0, stream>>>(Zht, Fre, Fim, V);
        tvnet_rowifft<<<gemmBlocks, 256, 0, stream>>>(V, Fre, Fim, uout);
        if (it + 1 < TVNITER)
            tvnet_prox<<<(NIMG6 * NN) / 256, 256, 0, stream>>>(uout, kerK,
                                                               gamma, p_, mu_);
    }
}
